// TorchAttention_78993038508116
// MI455X (gfx1250) — matrix operand order repaired and checked by reading, not compiled, not run on hardware
//
#include <hip/hip_runtime.h>
#include <hip/hip_bf16.h>

// ---------------------------------------------------------------------------
// Types for WMMA fragments (gfx1250, wave32)
// ---------------------------------------------------------------------------
typedef __attribute__((ext_vector_type(8)))  unsigned short us8;
typedef __attribute__((ext_vector_type(16))) __bf16         v16bf;
typedef __attribute__((ext_vector_type(8)))  float          v8f;

union Frag16 { us8 u[2]; v16bf v; };

#define ZERO8 {0.f,0.f,0.f,0.f,0.f,0.f,0.f,0.f}

__device__ __forceinline__ unsigned short f2bf(float f) {
    unsigned int u = __float_as_uint(f);
    u += 0x7FFFu + ((u >> 16) & 1u);          // round-to-nearest-even
    return (unsigned short)(u >> 16);
}

// ---------------------------------------------------------------------------
// Elementwise f32 -> bf16 convert
// ---------------------------------------------------------------------------
__global__ __launch_bounds__(256) void f32_to_bf16_kernel(
    const float* __restrict__ in, unsigned short* __restrict__ out, int n)
{
    for (int i = blockIdx.x * blockDim.x + threadIdx.x; i < n;
         i += gridDim.x * blockDim.x)
        out[i] = f2bf(in[i]);
}

// ---------------------------------------------------------------------------
// T5 relative-position bias LUT: lut[h][rel + 2047], rel = mem - ctx
// ---------------------------------------------------------------------------
__global__ __launch_bounds__(256) void bias_lut_kernel(
    const float* __restrict__ emb,   // [32, 16]
    float* __restrict__ lut)         // [16, 4096]
{
    int gid = blockIdx.x * 256 + threadIdx.x;
    if (gid >= 16 * 4096) return;
    int h = gid >> 12, idx = gid & 4095;
    int rel = idx - 2047;
    int rb = rel > 0 ? 16 : 0;                 // bidirectional, num_buckets 32->16
    int rp = rel < 0 ? -rel : rel;
    int bucket;
    if (rp < 8) {                              // max_exact = 8
        bucket = rp;
    } else {
        // 8 + log(rp/8)/log(128/8) * 8
        int lb = 8 + (int)(__logf((float)rp * 0.125f) * 2.885390082f);
        bucket = lb < 15 ? lb : 15;
    }
    lut[gid] = emb[(rb + bucket) * 16 + h];
}

// ---------------------------------------------------------------------------
// Transpose V: qkv[(b*S+s)*3072 + 2048 + h*64 + d] -> Vt[((b*16+h)*64+d)*S + s]
// ---------------------------------------------------------------------------
__global__ __launch_bounds__(256) void transpose_v_kernel(
    const unsigned short* __restrict__ qkv, unsigned short* __restrict__ Vt)
{
    int gid = blockIdx.x * 256 + threadIdx.x;     // over B*S*1024
    if (gid >= 2 * 2048 * 1024) return;
    int c = gid & 1023;
    int s = (gid >> 10) & 2047;
    int b = gid >> 21;
    int h = c >> 6, dcol = c & 63;
    unsigned short v = qkv[((size_t)(b * 2048 + s)) * 3072 + 2048 + c];
    Vt[(((size_t)(b * 16 + h)) * 64 + dcol) * 2048 + s] = v;
}

// ---------------------------------------------------------------------------
// Generic bf16 GEMM: C[M,N] = A[M,K] * B[N,K]^T   (A,B bf16 row-major)
// Block = 128 threads = 4 waves (2x2), each wave computes a 32x32 tile.
// ---------------------------------------------------------------------------
template<bool OUT_BF16>
__global__ __launch_bounds__(128) void gemm_bf16_kernel(
    const unsigned short* __restrict__ A,
    const unsigned short* __restrict__ Bw,
    void* __restrict__ C,
    int M, int N, int K)
{
    const int wave = threadIdx.x >> 5;
    const int lane = threadIdx.x & 31;
    const int lm = lane & 15;
    const int kg = lane >> 4;
    const int m0 = blockIdx.y * 64 + (wave >> 1) * 32;
    const int n0 = blockIdx.x * 64 + (wave & 1) * 32;

    const unsigned short* a0p = A  + (size_t)(m0 + lm) * K;
    const unsigned short* a1p = A  + (size_t)(m0 + 16 + lm) * K;
    const unsigned short* b0p = Bw + (size_t)(n0 + lm) * K;
    const unsigned short* b1p = Bw + (size_t)(n0 + 16 + lm) * K;

    v8f c00 = ZERO8, c01 = ZERO8, c10 = ZERO8, c11 = ZERO8;

    for (int k0 = 0; k0 < K; k0 += 32) {
        if (k0 + 128 < K) {                      // global_prefetch_b8
            __builtin_prefetch(a0p + k0 + 128, 0, 1);
            __builtin_prefetch(b0p + k0 + 128, 0, 1);
        }
        Frag16 a0, a1, b0, b1;
        a0.u[0] = *(const us8*)(a0p + k0 + kg * 8);
        a0.u[1] = *(const us8*)(a0p + k0 + 16 + kg * 8);
        a1.u[0] = *(const us8*)(a1p + k0 + kg * 8);
        a1.u[1] = *(const us8*)(a1p + k0 + 16 + kg * 8);
        b0.u[0] = *(const us8*)(b0p + k0 + kg * 8);
        b0.u[1] = *(const us8*)(b0p + k0 + 16 + kg * 8);
        b1.u[0] = *(const us8*)(b1p + k0 + kg * 8);
        b1.u[1] = *(const us8*)(b1p + k0 + 16 + kg * 8);
        c00 = __builtin_amdgcn_wmma_f32_16x16x32_bf16(false, a0.v, false, b0.v, (short)0, c00, false, false);
        c01 = __builtin_amdgcn_wmma_f32_16x16x32_bf16(false, a0.v, false, b1.v, (short)0, c01, false, false);
        c10 = __builtin_amdgcn_wmma_f32_16x16x32_bf16(false, a1.v, false, b0.v, (short)0, c10, false, false);
        c11 = __builtin_amdgcn_wmma_f32_16x16x32_bf16(false, a1.v, false, b1.v, (short)0, c11, false, false);
    }

    const int mb = kg * 8;                       // C layout: lanes16-31 -> rows +8
    #pragma unroll
    for (int j = 0; j < 8; ++j) {
        int mA = m0 + j + mb;
        int mB = m0 + 16 + j + mb;
        int nA = n0 + lm;
        int nB = n0 + 16 + lm;
        if (OUT_BF16) {
            unsigned short* o = (unsigned short*)C;
            o[(size_t)mA * N + nA] = f2bf(c00[j]);
            o[(size_t)mA * N + nB] = f2bf(c01[j]);
            o[(size_t)mB * N + nA] = f2bf(c10[j]);
            o[(size_t)mB * N + nB] = f2bf(c11[j]);
        } else {
            float* o = (float*)C;
            o[(size_t)mA * N + nA] = c00[j];
            o[(size_t)mA * N + nB] = c01[j];
            o[(size_t)mB * N + nA] = c10[j];
            o[(size_t)mB * N + nB] = c11[j];
        }
    }
}

// ---------------------------------------------------------------------------
// Flash attention: one wave handles 16 q-rows of one (b,h).
// qkv: [B*S, 3072] bf16 (Q|K|V), Vt: [B*16*64, S] bf16,
// mask: [B,1,S,S] f32, lut: [16,4096] f32, ctx out: [B*S, 1024] bf16.
// ---------------------------------------------------------------------------
__global__ __launch_bounds__(128) void attn_kernel(
    const unsigned short* __restrict__ qkv,
    const unsigned short* __restrict__ Vt,
    const float* __restrict__ mask,
    const float* __restrict__ lut,
    unsigned short* __restrict__ ctx)
{
    constexpr int S = 2048;
    __shared__ alignas(16) unsigned short plds[4][16][32];

    const int wave = threadIdx.x >> 5;
    const int lane = threadIdx.x & 31;
    const int lm = lane & 15;
    const int kg = lane >> 4;
    const int hb = kg * 8;

    const int tile = blockIdx.x * 4 + wave;      // 0..4095
    const int qt = tile & 127;
    const int h  = (tile >> 7) & 15;
    const int b  = tile >> 11;
    const int q0 = qt * 16;

    // Q fragments (A layout): rows q0..q0+15, d-halves [0,32) and [32,64)
    const unsigned short* qrow = qkv + ((size_t)(b * S + q0 + lm)) * 3072 + h * 64;
    Frag16 qa0, qa1;
    qa0.u[0] = *(const us8*)(qrow + kg * 8);
    qa0.u[1] = *(const us8*)(qrow + 16 + kg * 8);
    qa1.u[0] = *(const us8*)(qrow + 32 + kg * 8);
    qa1.u[1] = *(const us8*)(qrow + 48 + kg * 8);

    float mrow[8], lsum[8];
    v8f o[4] = {ZERO8, ZERO8, ZERO8, ZERO8};
    #pragma unroll
    for (int j = 0; j < 8; ++j) { mrow[j] = -3.0e38f; lsum[j] = 0.f; }

    const unsigned short* kbase = qkv + (size_t)(b * S) * 3072 + 1024 + h * 64;
    const unsigned short* vtb   = Vt + (((size_t)(b * 16 + h)) * 64) * S;
    const float* mrowp = mask + (size_t)b * S * S;
    const float* lutp  = lut + h * 4096;

    for (int kv0 = 0; kv0 < S; kv0 += 32) {
        float s[16];
        #pragma unroll
        for (int t = 0; t < 2; ++t) {
            const int cb = kv0 + t * 16;
            const unsigned short* kr = kbase + (size_t)(cb + lm) * 3072;
            Frag16 kb0, kb1;                     // B = K^T, d-halves 0..31/32..63
            kb0.u[0] = *(const us8*)(kr + kg * 8);
            kb0.u[1] = *(const us8*)(kr + 16 + kg * 8);
            kb1.u[0] = *(const us8*)(kr + 32 + kg * 8);
            kb1.u[1] = *(const us8*)(kr + 48 + kg * 8);
            v8f sc = ZERO8;
            sc = __builtin_amdgcn_wmma_f32_16x16x32_bf16(false, qa0.v, false, kb0.v, (short)0, sc, false, false);
            sc = __builtin_amdgcn_wmma_f32_16x16x32_bf16(false, qa1.v, false, kb1.v, (short)0, sc, false, false);
            const int ncol = cb + lm;
            #pragma unroll
            for (int j = 0; j < 8; ++j) {
                int qr = q0 + j + hb;
                float mv = mrowp[(size_t)qr * S + ncol];
                float bias = lutp[ncol - qr + 2047];
                s[t * 8 + j] = sc[j] * mv + (-1.0e8f * (1.0f - mv) + bias);
            }
        }
        // Online softmax update (row = j + hb, reduce over 16 lanes of each half)
        float scale[8];
        #pragma unroll
        for (int j = 0; j < 8; ++j) {
            float c = fmaxf(s[j], s[8 + j]);
            c = fmaxf(c, __shfl_xor(c, 1, 16));
            c = fmaxf(c, __shfl_xor(c, 2, 16));
            c = fmaxf(c, __shfl_xor(c, 4, 16));
            c = fmaxf(c, __shfl_xor(c, 8, 16));
            float mnew = fmaxf(mrow[j], c);
            scale[j] = __expf(mrow[j] - mnew);
            mrow[j] = mnew;
            float p0 = __expf(s[j] - mnew);
            float p1 = __expf(s[8 + j] - mnew);
            plds[wave][j + hb][lm]      = f2bf(p0);
            plds[wave][j + hb][16 + lm] = f2bf(p1);
            float ps = p0 + p1;
            ps += __shfl_xor(ps, 1, 16);
            ps += __shfl_xor(ps, 2, 16);
            ps += __shfl_xor(ps, 4, 16);
            ps += __shfl_xor(ps, 8, 16);
            lsum[j] = lsum[j] * scale[j] + ps;
        }
        // C-layout -> A-layout for P via per-wave LDS tile (same-wave DS in order)
        asm volatile("s_wait_dscnt 0" ::: "memory");
        Frag16 pa;
        pa.u[0] = *(const us8*)(&plds[wave][lm][kg * 8]);
        pa.u[1] = *(const us8*)(&plds[wave][lm][16 + kg * 8]);

        #pragma unroll
        for (int j = 0; j < 8; ++j) {
            o[0][j] *= scale[j]; o[1][j] *= scale[j];
            o[2][j] *= scale[j]; o[3][j] *= scale[j];
        }
        #pragma unroll
        for (int nt = 0; nt < 4; ++nt) {
            const unsigned short* vr = vtb + (size_t)(nt * 16 + lm) * S + kv0 + kg * 8;
            Frag16 vb;
            vb.u[0] = *(const us8*)(vr);
            vb.u[1] = *(const us8*)(vr + 16);
            o[nt] = __builtin_amdgcn_wmma_f32_16x16x32_bf16(false, pa.v, false, vb.v, (short)0, o[nt], false, false);
        }
    }

    #pragma unroll
    for (int j = 0; j < 8; ++j) {
        float inv = 1.0f / lsum[j];
        int qr = q0 + j + hb;
        unsigned short* crow = ctx + ((size_t)(b * S + qr)) * 1024 + h * 64 + lm;
        crow[0]  = f2bf(o[0][j] * inv);
        crow[16] = f2bf(o[1][j] * inv);
        crow[32] = f2bf(o[2][j] * inv);
        crow[48] = f2bf(o[3][j] * inv);
    }
}

// ---------------------------------------------------------------------------
// Launch
// ---------------------------------------------------------------------------
extern "C" void kernel_launch(void* const* d_in, const int* in_sizes, int n_in,
                              void* d_out, int out_size, void* d_ws, size_t ws_size,
                              hipStream_t stream)
{
    (void)in_sizes; (void)n_in; (void)out_size; (void)ws_size;
    const float* hs   = (const float*)d_in[0];   // [2,2048,1024]
    const float* am   = (const float*)d_in[1];   // [2,1,2048,2048]
    const float* wqkv = (const float*)d_in[2];   // [3072,1024]
    const float* remb = (const float*)d_in[3];   // [32,16]
    const float* wo   = (const float*)d_in[4];   // [1024,1024]
    float* out = (float*)d_out;                  // [2,2048,1024]

    char* ws = (char*)d_ws;
    unsigned short* Xb   = (unsigned short*)(ws);                          // 8 MiB
    unsigned short* Wqb  = (unsigned short*)(ws + ((size_t)8  << 20));     // 6 MiB
    unsigned short* Wob  = (unsigned short*)(ws + ((size_t)14 << 20));     // 2 MiB
    unsigned short* qkvb = (unsigned short*)(ws + ((size_t)16 << 20));     // 24 MiB
    unsigned short* Vtb  = (unsigned short*)(ws + ((size_t)40 << 20));     // 8 MiB
    unsigned short* ctxb = (unsigned short*)(ws + ((size_t)48 << 20));     // 8 MiB
    float* lut           = (float*)(ws + ((size_t)56 << 20));              // 256 KiB

    f32_to_bf16_kernel<<<2048, 256, 0, stream>>>(hs,   Xb,  2 * 2048 * 1024);
    f32_to_bf16_kernel<<<2048, 256, 0, stream>>>(wqkv, Wqb, 3072 * 1024);
    f32_to_bf16_kernel<<<1024, 256, 0, stream>>>(wo,   Wob, 1024 * 1024);
    bias_lut_kernel<<<256, 256, 0, stream>>>(remb, lut);

    // QKV projection: [4096,1024] x [3072,1024]^T -> bf16 [4096,3072]
    gemm_bf16_kernel<true><<<dim3(3072 / 64, 4096 / 64), 128, 0, stream>>>(
        Xb, Wqb, qkvb, 4096, 3072, 1024);

    transpose_v_kernel<<<(2 * 2048 * 1024) / 256, 256, 0, stream>>>(qkvb, Vtb);

    attn_kernel<<<1024, 128, 0, stream>>>(qkvb, Vtb, am, lut, ctxb);

    // Output projection: ctx [4096,1024] x Wo[1024,1024]^T -> f32 out
    gemm_bf16_kernel<false><<<dim3(1024 / 64, 4096 / 64), 128, 0, stream>>>(
        ctxb, Wob, out, 4096, 1024, 1024);
}
